// GCPNet_70952859730320
// MI455X (gfx1250) — matrix operand order repaired and checked by reading, not compiled, not run on hardware
//
#include <hip/hip_runtime.h>
#include <hip/hip_bf16.h>

#define EMB 64
#define NVERT 4096
#define NCON 10
#define TITER 32

typedef __attribute__((ext_vector_type(16))) _Float16 v16h;
typedef __attribute__((ext_vector_type(8)))  _Float16 v8h;
typedef __attribute__((ext_vector_type(8)))  float    v8f;

// ---------------- WMMA fragment helpers (CDNA5 16x16x32 f16, wave32) -------------

// A fragment: 16x32 f16 tile, row-major source, lda in halfs.
// lane<16: row=lane, K = 0..7 (elems 0..7) and 16..23 (elems 8..15)
// lane>=16: row=lane-16, K = 8..15 and 24..31
__device__ __forceinline__ v16h gcp_load_a(const _Float16* tile, int lda) {
    int lane = threadIdx.x & 31;
    int row  = lane & 15;
    int kh   = (lane >> 4) << 3;
    const _Float16* p = tile + row * lda + kh;
    v8h lo = *(const v8h*)p;
    v8h hi = *(const v8h*)(p + 16);
    v16h a;
#pragma unroll
    for (int t = 0; t < 8; ++t) { a[t] = lo[t]; a[8 + t] = hi[t]; }
    return a;
}

// B fragment: 32x16 f16 tile from a COLUMN-MAJOR matrix (column stride ldb halfs).
// lane<16: col=lane, K = 0..7 and 16..23; lane>=16: col=lane-16, K = 8..15 and 24..31.
__device__ __forceinline__ v16h gcp_load_b(const _Float16* tile, int ldb) {
    int lane = threadIdx.x & 31;
    int col  = lane & 15;
    int ko   = (lane >> 4) << 3;
    const _Float16* p = tile + col * ldb + ko;
    v8h lo = *(const v8h*)p;
    v8h hi = *(const v8h*)(p + 16);
    v16h b;
#pragma unroll
    for (int t = 0; t < 8; ++t) { b[t] = lo[t]; b[8 + t] = hi[t]; }
    return b;
}

__device__ __forceinline__ v8f gcp_wmma(v16h a, v16h b, v8f c) {
    return __builtin_amdgcn_wmma_f32_16x16x32_f16(false, a, false, b, (short)0, c,
                                                  false, false);
}

__device__ __forceinline__ float gcp_sigmoid(float x) {
    return 1.0f / (1.0f + __expf(-x));
}

// ---------------- prep / pack kernels --------------------------------------------

__global__ void gcp_cvt_f16(const float* __restrict__ src, _Float16* __restrict__ dst,
                            int n) {
    for (int i = blockIdx.x * blockDim.x + threadIdx.x; i < n;
         i += gridDim.x * blockDim.x)
        dst[i] = (_Float16)src[i];
}

// out (col-major f16, K x Nn) from in (row-major f32). out[n*K+k] = in[k*Nn+n]
__global__ void gcp_pack_cm(const float* __restrict__ in, _Float16* __restrict__ out,
                            int K, int Nn) {
    int tot = K * Nn;
    for (int i = blockIdx.x * blockDim.x + threadIdx.x; i < tot;
         i += gridDim.x * blockDim.x) {
        int n = i / K, k = i % K;
        out[i] = (_Float16)in[k * Nn + n];
    }
}

__global__ void gcp_init(const float* __restrict__ V0, const float* __restrict__ C0,
                         const float* __restrict__ Vh0, const float* __restrict__ Ch0,
                         float* Vf, float* Cf, float* vh, float* vc, float* ch,
                         float* cc) {
    for (int i = blockIdx.x * blockDim.x + threadIdx.x; i < NVERT * EMB;
         i += gridDim.x * blockDim.x) {
        Vf[i] = V0[i];
        if (i < NCON * EMB) Cf[i] = C0[i];
        if (i < EMB) { vh[i] = Vh0[i]; vc[i] = 0.f; ch[i] = Ch0[i]; cc[i] = 0.f; }
    }
}

// V f32 [N,E] -> Vh f16 row-major and VT f16 [E,N] (column-contiguous B operand)
__global__ void gcp_prep_v(const float* __restrict__ Vf, _Float16* __restrict__ Vh,
                           _Float16* __restrict__ VT) {
    for (int i = blockIdx.x * blockDim.x + threadIdx.x; i < NVERT * EMB;
         i += gridDim.x * blockDim.x) {
        int n = i >> 6, e = i & 63;
        _Float16 h = (_Float16)Vf[i];
        Vh[i] = h;
        VT[e * NVERT + n] = h;
    }
}

// ---------------- fused 2/3-layer 64-wide MLP via WMMA ---------------------------
// X f16 [N,64] row-major; weights f16 column-major [64x64]; out f32 [N,64].
// Compile-time layer count / relu so the tile epilogue is branch-free.
template <int NLAYERS, bool RELU_LAST>
__global__ __launch_bounds__(128) void gcp_mlp64_wmma(
    const _Float16* __restrict__ Xh, const _Float16* W0, const float* b0,
    const _Float16* W1, const float* b1, const _Float16* W2, const float* b2,
    float* __restrict__ out) {
    __shared__ _Float16 buf0[64 * 72];
    __shared__ _Float16 buf1[64 * 72];
    const int w = threadIdx.x >> 5, lane = threadIdx.x & 31;
    const int col = lane & 15, rb = (lane >> 4) << 3;
    const int mbase = blockIdx.x * 64;
    const _Float16* Ws[3] = {W0, W1, W2};
    const float* Bs[3] = {b0, b1, b2};

#pragma unroll
    for (int L = 0; L < NLAYERS; ++L) {
        const _Float16* src;
        int lda;
        if (L == 0) { src = Xh + (size_t)mbase * EMB; lda = EMB; }
        else        { src = (L & 1) ? buf0 : buf1;    lda = 72;  }
        _Float16* dst = (L & 1) ? buf1 : buf0;
        const bool last = (L == NLAYERS - 1);
        const bool relu = (L < NLAYERS - 1) || RELU_LAST;
#pragma unroll
        for (int nsub = 0; nsub < 4; ++nsub) {
            v8f acc = {};
#pragma unroll
            for (int ks = 0; ks < 2; ++ks) {
                v16h a = gcp_load_a(src + (w * 16) * lda + ks * 32, lda);
                v16h b = gcp_load_b(Ws[L] + (nsub * 16) * EMB + ks * 32, EMB);
                acc = gcp_wmma(a, b, acc);
            }
            float bias = Bs[L][nsub * 16 + col];
#pragma unroll
            for (int r = 0; r < 8; ++r) {
                float v = acc[r] + bias;
                if (relu) v = fmaxf(v, 0.f);
                int row = w * 16 + rb + r;
                if (last) out[(size_t)(mbase + row) * EMB + nsub * 16 + col] = v;
                else      dst[row * 72 + nsub * 16 + col] = (_Float16)v;
            }
        }
        __syncthreads();
    }
}

// ---------------- big adjacency GEMM: lstm_in[:,0:64] = f16(M_vv @ V) ------------
// A = M_vv f16 [N,N] row-major; B = VT f16 [64,N] column-contiguous.
__global__ __launch_bounds__(128) void gcp_mv_wmma(const _Float16* __restrict__ Ah,
                                                   const _Float16* __restrict__ BT,
                                                   _Float16* __restrict__ lin) {
    __shared__ _Float16 at[64 * 40];  // 64 rows x 32 halfs, padded stride 40
    const int w = threadIdx.x >> 5, lane = threadIdx.x & 31;
    const int col = lane & 15, rb = (lane >> 4) << 3;
    const int mbase = blockIdx.x * 64;
    const int r0 = threadIdx.x >> 1, seg = threadIdx.x & 1;  // cooperative copy role
    v8f acc[4] = {{}, {}, {}, {}};

    const _Float16* arow = Ah + (size_t)(mbase + r0) * NVERT + seg * 16;
    for (int ks = 0; ks < NVERT / 32; ++ks) {
        const _Float16* gp = arow + ks * 32;
        _Float16* sp = at + r0 * 40 + seg * 16;
        *(v8h*)sp = *(const v8h*)gp;
        *(v8h*)(sp + 8) = *(const v8h*)(gp + 8);
        if (ks + 1 < NVERT / 32)
            __builtin_prefetch(gp + 32, 0, 1);  // global_prefetch_b8: next K tile
        __syncthreads();
        v16h a = gcp_load_a(at + (w * 16) * 40, 40);
#pragma unroll
        for (int nsub = 0; nsub < 4; ++nsub) {
            v16h b = gcp_load_b(BT + (size_t)(nsub * 16) * NVERT + ks * 32, NVERT);
            acc[nsub] = gcp_wmma(a, b, acc[nsub]);
        }
        __syncthreads();
    }
#pragma unroll
    for (int nsub = 0; nsub < 4; ++nsub)
#pragma unroll
        for (int r = 0; r < 8; ++r)
            lin[(size_t)(mbase + w * 16 + rb + r) * 128 + nsub * 16 + col] =
                (_Float16)acc[nsub][r];
}

// lstm_in[:,64:128] = f16(M_vc @ mlp_C)   (K = 10, scalar path)
__global__ void gcp_mc(const float* __restrict__ Mvc, const float* __restrict__ mlpC,
                       _Float16* __restrict__ lin) {
    for (int i = blockIdx.x * blockDim.x + threadIdx.x; i < NVERT * EMB;
         i += gridDim.x * blockDim.x) {
        int n = i >> 6, e = i & 63;
        float acc = 0.f;
#pragma unroll
        for (int c = 0; c < NCON; ++c) acc += Mvc[n * NCON + c] * mlpC[c * EMB + e];
        lin[(size_t)n * 128 + 64 + e] = (_Float16)acc;
    }
}

// xz = lstm_in @ lv_Wih + (bih+bhh);  A f16 [N,128], B f16 cm [256 cols][128]
__global__ __launch_bounds__(128) void gcp_xz_wmma(const _Float16* __restrict__ Xh,
                                                   const _Float16* __restrict__ Wcm,
                                                   const float* __restrict__ bih,
                                                   const float* __restrict__ bhh,
                                                   float* __restrict__ xz) {
    __shared__ _Float16 xt[64 * 136];  // 64 rows x 128 halfs, padded stride 136
    const int w = threadIdx.x >> 5, lane = threadIdx.x & 31;
    const int col = lane & 15, rb = (lane >> 4) << 3;
    const int mbase = blockIdx.x * 64;
    const int r0 = threadIdx.x >> 1, seg = threadIdx.x & 1;
    {
        const _Float16* gp = Xh + (size_t)(mbase + r0) * 128 + seg * 64;
        _Float16* sp = xt + r0 * 136 + seg * 64;
#pragma unroll
        for (int q = 0; q < 8; ++q) *(v8h*)(sp + q * 8) = *(const v8h*)(gp + q * 8);
    }
    __syncthreads();
    v16h af[4];
#pragma unroll
    for (int ks = 0; ks < 4; ++ks)
        af[ks] = gcp_load_a(xt + (w * 16) * 136 + ks * 32, 136);
    for (int nsub = 0; nsub < 16; ++nsub) {
        v8f acc = {};
#pragma unroll
        for (int ks = 0; ks < 4; ++ks) {
            v16h b = gcp_load_b(Wcm + (nsub * 16) * 128 + ks * 32, 128);
            acc = gcp_wmma(af[ks], b, acc);
        }
        int n = nsub * 16 + col;
        float bias = bih[n] + bhh[n];
#pragma unroll
        for (int r = 0; r < 8; ++r)
            xz[(size_t)(mbase + w * 16 + rb + r) * 256 + n] = acc[r] + bias;
    }
}

// ---------------- tiny constraint-side kernels -----------------------------------

// 3-layer MLP on C [10,64], f32 scalar (640 outputs/layer)
__global__ void gcp_mlpc(const float* __restrict__ Cf, const float* W0,
                         const float* W1, const float* W2, const float* b0,
                         const float* b1, const float* b2,
                         float* __restrict__ out) {
    __shared__ float xa[NCON * EMB], xb[NCON * EMB];
    const float* Ws[3] = {W0, W1, W2};
    const float* Bs[3] = {b0, b1, b2};
    for (int j = threadIdx.x; j < NCON * EMB; j += blockDim.x) xa[j] = Cf[j];
    __syncthreads();
    float* cur = xa; float* nxt = xb;
    for (int L = 0; L < 3; ++L) {
        for (int j = threadIdx.x; j < NCON * EMB; j += blockDim.x) {
            int row = j >> 6, e = j & 63;
            float acc = Bs[L][e];
            for (int k = 0; k < EMB; ++k) acc += cur[row * EMB + k] * Ws[L][k * EMB + e];
            if (L < 2) acc = fmaxf(acc, 0.f);
            if (L == 2) out[j] = acc; else nxt[j] = acc;
        }
        __syncthreads();
        float* t = cur; cur = nxt; nxt = t;
    }
}

// CT = M_vc^T @ mlp_V   ([10,64], K = 4096 block reduction)
__global__ void gcp_ct(const float* __restrict__ Mvc, const float* __restrict__ mlpV,
                       float* __restrict__ CT) {
    __shared__ float red[256];
    int c = blockIdx.x;
    int part = threadIdx.x >> 6, e = threadIdx.x & 63;
    float acc = 0.f;
    for (int n = part; n < NVERT; n += 4)
        acc += Mvc[n * NCON + c] * mlpV[(size_t)n * EMB + e];
    red[threadIdx.x] = acc;
    __syncthreads();
    if (part == 0)
        CT[c * EMB + e] = red[e] + red[64 + e] + red[128 + e] + red[192 + e];
}

// cxz = CT @ lc_Wih + (bih+bhh)   ([10,256])
__global__ void gcp_cxz(const float* __restrict__ CT, const float* __restrict__ Wih,
                        const float* __restrict__ bih, const float* __restrict__ bhh,
                        float* __restrict__ cxz) {
    int j = threadIdx.x;
    for (int c = 0; c < NCON; ++c) {
        float acc = bih[j] + bhh[j];
        for (int k = 0; k < EMB; ++k) acc += CT[c * EMB + k] * Wih[k * 256 + j];
        cxz[c * 256 + j] = acc;
    }
}

// ---------------- persistent single-workgroup LSTM scan --------------------------
// xz [T,256] precomputed; Whh [64,256] row-major; thread j owns gate output j and
// keeps Whh[:,j] in 64 VGPRs. h,c,z live in LDS; two barriers per step.
__global__ __launch_bounds__(256) void gcp_lstm_scan(const float* __restrict__ xz,
                                                     int T,
                                                     const float* __restrict__ Whh,
                                                     float* hstate, float* cstate,
                                                     float* __restrict__ Y) {
    __shared__ float hs[EMB], cs[EMB], zs[256];
    const int j = threadIdx.x;
    float wc[EMB];
#pragma unroll
    for (int k = 0; k < EMB; ++k) wc[k] = Whh[k * 256 + j];
    if (j < EMB) { hs[j] = hstate[j]; cs[j] = cstate[j]; }
    __syncthreads();
    for (int t = 0; t < T; ++t) {
        float a0 = 0.f, a1 = 0.f, a2 = 0.f, a3 = 0.f;
#pragma unroll
        for (int k = 0; k < EMB; k += 4) {
            a0 += hs[k] * wc[k];         a1 += hs[k + 1] * wc[k + 1];
            a2 += hs[k + 2] * wc[k + 2]; a3 += hs[k + 3] * wc[k + 3];
        }
        zs[j] = xz[(size_t)t * 256 + j] + ((a0 + a1) + (a2 + a3));
        __syncthreads();
        if (j < EMB) {
            float ig = zs[j], fg = zs[64 + j], gg = zs[128 + j], og = zs[192 + j];
            float c2 = gcp_sigmoid(fg) * cs[j] + gcp_sigmoid(ig) * tanhf(gg);
            float h2 = gcp_sigmoid(og) * tanhf(c2);
            cs[j] = c2; hs[j] = h2;
            Y[(size_t)t * EMB + j] = h2;
        }
        __syncthreads();
    }
    if (j < EMB) { hstate[j] = hs[j]; cstate[j] = cs[j]; }
}

// ---------------- vote tail: dot with w3, per-graph mean, sigmoid ----------------
__global__ __launch_bounds__(512) void gcp_vote_out(const float* __restrict__ X,
                                                    const float* __restrict__ w3,
                                                    const float* __restrict__ b3,
                                                    float* __restrict__ out) {
    __shared__ float red[512];
    int n = blockIdx.x * 512 + threadIdx.x;
    float d = 0.f;
#pragma unroll 8
    for (int e = 0; e < EMB; ++e) d += X[(size_t)n * EMB + e] * w3[e];
    red[threadIdx.x] = d;
    __syncthreads();
    for (int s = 256; s > 0; s >>= 1) {
        if (threadIdx.x < s) red[threadIdx.x] += red[threadIdx.x + s];
        __syncthreads();
    }
    if (threadIdx.x == 0)
        out[blockIdx.x] = gcp_sigmoid(red[0] / 512.f + b3[0]);
}

// ---------------- host orchestration ---------------------------------------------

extern "C" void kernel_launch(void* const* d_in, const int* in_sizes, int n_in,
                              void* d_out, int out_size, void* d_ws, size_t ws_size,
                              hipStream_t stream) {
    (void)in_sizes; (void)n_in; (void)ws_size;
    const float* Mvv   = (const float*)d_in[0];
    const float* Mvc   = (const float*)d_in[1];
    const float* V0    = (const float*)d_in[2];
    const float* C0    = (const float*)d_in[3];
    const float* mVW[3] = {(const float*)d_in[4], (const float*)d_in[5], (const float*)d_in[6]};
    const float* mVb[3] = {(const float*)d_in[7], (const float*)d_in[8], (const float*)d_in[9]};
    const float* mCW[3] = {(const float*)d_in[10], (const float*)d_in[11], (const float*)d_in[12]};
    const float* mCb[3] = {(const float*)d_in[13], (const float*)d_in[14], (const float*)d_in[15]};
    const float* vW[3]  = {(const float*)d_in[16], (const float*)d_in[17], (const float*)d_in[18]};
    const float* vb[3]  = {(const float*)d_in[19], (const float*)d_in[20], (const float*)d_in[21]};
    const float* lvWih = (const float*)d_in[22];
    const float* lvWhh = (const float*)d_in[23];
    const float* lvbih = (const float*)d_in[24];
    const float* lvbhh = (const float*)d_in[25];
    const float* lcWih = (const float*)d_in[26];
    const float* lcWhh = (const float*)d_in[27];
    const float* lcbih = (const float*)d_in[28];
    const float* lcbhh = (const float*)d_in[29];
    const float* Vh0   = (const float*)d_in[30];
    const float* Ch0   = (const float*)d_in[31];

    // bump allocator inside d_ws
    char* wp = (char*)d_ws;
    auto take = [&](size_t bytes) {
        void* r = (void*)wp;
        wp += (bytes + 255) & ~(size_t)255;
        return r;
    };
    _Float16* Avvh  = (_Float16*)take((size_t)NVERT * NVERT * 2);  // 32 MB, L2-resident
    _Float16* Vh    = (_Float16*)take((size_t)NVERT * EMB * 2);
    _Float16* VT    = (_Float16*)take((size_t)NVERT * EMB * 2);
    float*    Vf    = (float*)take((size_t)NVERT * EMB * 4);
    float*    Cf    = (float*)take((size_t)NCON * EMB * 4);
    float*    mlpVo = (float*)take((size_t)NVERT * EMB * 4);
    float*    mlpCo = (float*)take((size_t)NCON * EMB * 4);
    _Float16* LIh   = (_Float16*)take((size_t)NVERT * 128 * 2);
    float*    XZ    = (float*)take((size_t)NVERT * 256 * 4);
    float*    CTb   = (float*)take((size_t)NCON * EMB * 4);
    float*    CXZ   = (float*)take((size_t)NCON * 256 * 4);
    float*    vh    = (float*)take(EMB * 4);
    float*    vc    = (float*)take(EMB * 4);
    float*    ch    = (float*)take(EMB * 4);
    float*    cc    = (float*)take(EMB * 4);
    _Float16* W0cm  = (_Float16*)take(EMB * EMB * 2);
    _Float16* W1cm  = (_Float16*)take(EMB * EMB * 2);
    _Float16* W2cm  = (_Float16*)take(EMB * EMB * 2);
    _Float16* vW0cm = (_Float16*)take(EMB * EMB * 2);
    _Float16* vW1cm = (_Float16*)take(EMB * EMB * 2);
    _Float16* Wihcm = (_Float16*)take(128 * 256 * 2);
    float*    votex = (float*)take((size_t)NVERT * EMB * 4);

    // one-time per launch: f16 adjacency + column-major f16 weights + state init
    gcp_cvt_f16<<<2048, 256, 0, stream>>>(Mvv, Avvh, NVERT * NVERT);
    gcp_pack_cm<<<16, 256, 0, stream>>>(mVW[0], W0cm, EMB, EMB);
    gcp_pack_cm<<<16, 256, 0, stream>>>(mVW[1], W1cm, EMB, EMB);
    gcp_pack_cm<<<16, 256, 0, stream>>>(mVW[2], W2cm, EMB, EMB);
    gcp_pack_cm<<<16, 256, 0, stream>>>(vW[0], vW0cm, EMB, EMB);
    gcp_pack_cm<<<16, 256, 0, stream>>>(vW[1], vW1cm, EMB, EMB);
    gcp_pack_cm<<<128, 256, 0, stream>>>(lvWih, Wihcm, 128, 256);
    gcp_init<<<1024, 256, 0, stream>>>(V0, C0, Vh0, Ch0, Vf, Cf, vh, vc, ch, cc);

    for (int it = 0; it < TITER; ++it) {
        gcp_prep_v<<<512, 256, 0, stream>>>(Vf, Vh, VT);
        gcp_mlp64_wmma<3, false><<<NVERT / 64, 128, 0, stream>>>(
            Vh, W0cm, mVb[0], W1cm, mVb[1], W2cm, mVb[2], mlpVo);
        gcp_mlpc<<<1, 256, 0, stream>>>(Cf, mCW[0], mCW[1], mCW[2], mCb[0], mCb[1],
                                        mCb[2], mlpCo);
        gcp_mc<<<1024, 256, 0, stream>>>(Mvc, mlpCo, LIh);
        gcp_mv_wmma<<<NVERT / 64, 128, 0, stream>>>(Avvh, VT, LIh);
        gcp_xz_wmma<<<NVERT / 64, 128, 0, stream>>>(LIh, Wihcm, lvbih, lvbhh, XZ);
        gcp_ct<<<NCON, 256, 0, stream>>>(Mvc, mlpVo, CTb);
        gcp_cxz<<<1, 256, 0, stream>>>(CTb, lcWih, lcbih, lcbhh, CXZ);
        gcp_lstm_scan<<<1, 256, 0, stream>>>(XZ, NVERT, lvWhh, vh, vc, Vf);
        gcp_lstm_scan<<<1, 256, 0, stream>>>(CXZ, NCON, lcWhh, ch, cc, Cf);
    }

    // vote head: two 64x64 ReLU layers via WMMA, then dot+mean+sigmoid
    gcp_prep_v<<<512, 256, 0, stream>>>(Vf, Vh, VT);
    gcp_mlp64_wmma<2, true><<<NVERT / 64, 128, 0, stream>>>(
        Vh, vW0cm, vb[0], vW1cm, vb[1], nullptr, nullptr, votex);
    gcp_vote_out<<<out_size, 512, 0, stream>>>(votex, vW[2], vb[2], (float*)d_out);
}
